// Model_79929341379019
// MI455X (gfx1250) — matrix operand order repaired and checked by reading, not compiled, not run on hardware
//
#include <hip/hip_runtime.h>
#include <hip/hip_bf16.h>

typedef _Float16 half_t;
typedef __attribute__((ext_vector_type(16))) _Float16 v16h;
typedef __attribute__((ext_vector_type(8)))  _Float16 v8h;
typedef __attribute__((ext_vector_type(8)))  float    v8f;
typedef __attribute__((ext_vector_type(4)))  unsigned int v4u;
typedef __attribute__((ext_vector_type(8)))  int      v8i;
typedef __attribute__((ext_vector_type(4)))  int      v4i;

// ---------------------------------------------------------------------------
// CDNA5 Tensor Data Mover support (probe-verified builtin; arity differs by
// toolchain: ROCm 7.2 clang-22 -> 5 args, therock clang-23 -> 6 args).
// ---------------------------------------------------------------------------
#if defined(__has_builtin)
#  if __has_builtin(__builtin_amdgcn_tensor_load_to_lds)
#    define HAVE_TDM 1
#  endif
#endif
#ifndef HAVE_TDM
#  define HAVE_TDM 0
#endif
#if HAVE_TDM && __has_include(<hip/amd_detail/amd_gfx1250_TDM.h>)
#  define TDM_ARGS6 1
#else
#  define TDM_ARGS6 0
#endif

__device__ __forceinline__ unsigned lds_off32(const void* p) {
  // Flat LDS aperture: byte offset within wave's LDS lives in addr[31:0].
  return (unsigned)(unsigned long long)p;
}

#if HAVE_TDM
// 2D tile load: tile_d0 (contiguous, halves) x tile_d1 rows, source row
// stride = stride_elems halves. Destination LDS rows padded: 16 DWORDs of
// data then 4 DWORDs pad  => LDS row stride of 40 halves for tile_d0 == 32.
__device__ __forceinline__ void tdm_load_2d(unsigned lds_addr, const void* gptr,
                                            int tile_d0, int tile_d1,
                                            int stride_elems) {
  const unsigned long long ga = (unsigned long long)gptr;
  v4u g0;
  g0.x = 1u;                                   // count = 1 valid descriptor
  g0.y = lds_addr;                             // LDS byte address
  g0.z = (unsigned)ga;                         // global_addr[31:0]
  g0.w = (unsigned)((ga >> 32) & 0x01FFFFFFu)  // global_addr[56:32]
       | (2u << 30);                           // type = 2 ("image")
  unsigned long long q0 = 0, q1 = 0, q2 = 0, q3 = 0;
  q0 |= (1ull << 16);                          // data_size = 2 bytes
  q0 |= (1ull << 20);                          // pad_enable
  q0 |= (3ull << 22);                          // pad_interval: 16 DWORDs
  q0 |= (3ull << 25);                          // pad_amount:   4 DWORDs
  const unsigned long long td0 = 0x40000000ull;      // tensor_dim0 (no OOB)
  const unsigned long long td1 = 0x40000000ull;      // tensor_dim1 (no OOB)
  q0 |= (td0 & 0xFFFFull) << 48;                     // tensor_dim0[15:0]
  q1 |= (td0 >> 16) & 0xFFFFull;                     // tensor_dim0[31:16]
  q1 |= (td1 & 0xFFFFFFFFull) << 16;                 // tensor_dim1
  q1 |= ((unsigned long long)(unsigned)tile_d0) << 48;   // tile_dim0
  q2 |= (unsigned long long)(unsigned)tile_d1;           // tile_dim1
  const unsigned long long st0 = (unsigned long long)(unsigned)stride_elems;
  q2 |= (st0 & 0xFFFFFFFFull) << 32;                 // tensor_dim0_stride lo
  q3 |= (st0 >> 32) & 0xFFFFull;                     // tensor_dim0_stride hi
  v8i g1;
  g1[0] = (int)(unsigned)q0; g1[1] = (int)(unsigned)(q0 >> 32);
  g1[2] = (int)(unsigned)q1; g1[3] = (int)(unsigned)(q1 >> 32);
  g1[4] = (int)(unsigned)q2; g1[5] = (int)(unsigned)(q2 >> 32);
  g1[6] = (int)(unsigned)q3; g1[7] = (int)(unsigned)(q3 >> 32);
  const v4i z4 = {0, 0, 0, 0};
#if TDM_ARGS6
  const v8i z8 = {0, 0, 0, 0, 0, 0, 0, 0};
  __builtin_amdgcn_tensor_load_to_lds(g0, g1, z4, z4, z8, 0);
#else
  __builtin_amdgcn_tensor_load_to_lds(g0, g1, z4, z4, 0);
#endif
}
#endif

template<int N>
__device__ __forceinline__ void tdm_wait() {
#if defined(__has_builtin) && __has_builtin(__builtin_amdgcn_s_wait_tensorcnt)
  __builtin_amdgcn_s_wait_tensorcnt(N);
#else
  asm volatile("s_wait_tensorcnt %0" :: "n"(N) : "memory");
#endif
}

// ---------------------------------------------------------------------------
// Generic WMMA GEMM: C[M,N] = act( alpha * A[M,K] @ B[K,N] + bias [+ resid] )
// A row-major f16 (lda). B row-major f16 (ldb), or if TRANS_B, B is (N,K)
// row-major (QK^T scores). Block = 256 threads = 8 waves, tile 128 x (SN*32);
// wave grid 4(m) x 2(n), per-wave 32 x SN*16 patch = 2 x SN WMMA subtiles.
// Ping-pong LDS double buffering: the Tensor Data Mover streams K-slice t+1
// while the waves run WMMAs on K-slice t; TENSORcnt's in-order completion
// lets wave 0 wait "<= NDESC" to retire tile t while tile t+1 stays in
// flight behind the compute phase.
// ---------------------------------------------------------------------------
struct GemmP {
  const half_t* A;
  const half_t* B;
  const float*  bias;   // per-column bias (N), optional
  float*        resid;  // f32 residual stream, read+write in place, optional
  half_t*       outH;   // f16 output, optional
  float*        outF;   // f32 output, optional
  const unsigned char* mask; // (B,S) attention mask, optional
  int M, N, K;
  int lda, ldb, ldc;
  int zH;               // z = bz*zH + hz decomposition for batched attention
  long aZb, aZh, bZb, bZh, cZb, cZh;
  float alpha;
};

template<int SN, bool TRANS_B, bool OUT_HALF, bool BIAS, bool RELU,
         bool RESIDUAL, bool MASKED>
__global__ __launch_bounds__(256) void gemm_wmma(GemmP p) {
  constexpr int BN    = SN * 32;          // block N-tile
  constexpr int NDESC = TRANS_B ? 2 : 1;  // TDM descriptors per K-slice
  const int z  = blockIdx.z;
  const int bz = z / p.zH;
  const int hz = z % p.zH;
  const long aOff = (long)bz * p.aZb + (long)hz * p.aZh;
  const long bOff = (long)bz * p.bZb + (long)hz * p.bZh;
  const long cOff = (long)bz * p.cZb + (long)hz * p.cZh;
  const half_t* __restrict__ A = p.A + aOff;
  const half_t* __restrict__ B = p.B + bOff;

  const int m0 = blockIdx.y * 128;
  const int n0 = blockIdx.x * BN;

  __shared__ __align__(16) half_t As[2][128 * 40]; // 128 rows x 32 K, str 40
  __shared__ __align__(16) half_t Bs[2][BN * 40];  //  BN cols x 32 K, str 40

  const int tid  = threadIdx.x;
  const int lane = tid & 31;
  const int w    = tid >> 5;      // wave id 0..7
  const int wm   = w >> 1;        // 0..3 (M direction)
  const int wn   = w & 1;         // 0..1 (N direction)
  const int l16  = lane & 15;
  const int hs   = lane >> 4;     // lane half-select per ISA WMMA layout

  // Issue all loads that stage K-slice starting at element kk into buf.
  auto stage = [&](int kk, int buf) {
#if HAVE_TDM
    if (w == 0) {                 // one wave issues the TDM descriptors
      tdm_load_2d(lds_off32(&As[buf][0]), A + (long)m0 * p.lda + kk,
                  32, 128, p.lda);
      if (TRANS_B)
        tdm_load_2d(lds_off32(&Bs[buf][0]), B + (long)n0 * p.ldb + kk,
                    32, BN, p.ldb);
    }
#else
    #pragma unroll
    for (int i = 0; i < 2; ++i) { // A tile: 128x32 halves, 16B chunks
      int c   = tid + i * 256;
      int row = c >> 2;
      int col = (c & 3) << 3;
      *(uint4*)(&As[buf][row * 40 + col]) =
          *(const uint4*)(A + (long)(m0 + row) * p.lda + kk + col);
    }
    if (TRANS_B) {
      #pragma unroll
      for (int c0 = 0; c0 < BN * 4; c0 += 256) {
        int c    = c0 + tid;
        int nrow = c >> 2;
        int kcol = (c & 3) << 3;
        *(uint4*)(&Bs[buf][nrow * 40 + kcol]) =
            *(const uint4*)(B + (long)(n0 + nrow) * p.ldb + kk + kcol);
      }
    }
#endif
    if (!TRANS_B) {               // B tile: transpose-on-store into Bs[n][k]
      #pragma unroll
      for (int c0 = 0; c0 < 32 * (BN / 8); c0 += 256) {
        int c    = c0 + tid;
        int krow = c / (BN / 8);
        int ncol = (c % (BN / 8)) << 3;
        union { uint4 u; half_t h[8]; } t;
        t.u = *(const uint4*)(B + (long)(kk + krow) * p.ldb + n0 + ncol);
        #pragma unroll
        for (int j = 0; j < 8; ++j) Bs[buf][(ncol + j) * 40 + krow] = t.h[j];
      }
    }
  };

  v8f acc[2][SN] = {};
  const int nT = p.K >> 5;        // number of 32-deep K-slices

  stage(0, 0);                    // prologue: start tile 0
  for (int t = 0; t < nT; ++t) {
    const int cur = t & 1;
    __syncthreads();              // all waves done reading buffer cur^1
    if (t + 1 < nT) stage((t + 1) << 5, cur ^ 1);   // prefetch next slice
#if HAVE_TDM
    if (w == 0) {
      if (t + 1 < nT) tdm_wait<NDESC>();  // tile t done, tile t+1 in flight
      else            tdm_wait<0>();      // drain
    }
#endif
    __syncthreads();              // publish tile t to all waves

    // ---- fragments per ISA 16-bit A(16x32)/B(32x16) VGPR layouts ----------
    v16h af[2], bf[SN];
    #pragma unroll
    for (int sm = 0; sm < 2; ++sm) {
      int m = wm * 32 + sm * 16 + l16;
      union { v16h v; struct { v8h lo, hi; } s; } u;
      u.s.lo = *(const v8h*)(&As[cur][m * 40 + hs * 8]);      // K hs*8..+7
      u.s.hi = *(const v8h*)(&As[cur][m * 40 + hs * 8 + 16]); // K 16+hs*8..
      af[sm] = u.v;
    }
    #pragma unroll
    for (int sn = 0; sn < SN; ++sn) {
      int n = wn * (SN * 16) + sn * 16 + l16;
      union { v16h v; struct { v8h lo, hi; } s; } u;
      u.s.lo = *(const v8h*)(&Bs[cur][n * 40 + hs * 8]);      // K hs*8..+7
      u.s.hi = *(const v8h*)(&Bs[cur][n * 40 + hs * 8 + 16]); // K 16+hs*8..
      bf[sn] = u.v;
    }
    #pragma unroll
    for (int sm = 0; sm < 2; ++sm)
      #pragma unroll
      for (int sn = 0; sn < SN; ++sn)
        acc[sm][sn] = __builtin_amdgcn_wmma_f32_16x16x32_f16(
            false, af[sm], false, bf[sn], (short)0, acc[sm][sn], false, false);
  }

  // ---- epilogue: C layout VGPR r -> M = base + hs*8 + r, N = l16 ----------
  #pragma unroll
  for (int sm = 0; sm < 2; ++sm) {
    #pragma unroll
    for (int sn = 0; sn < SN; ++sn) {
      const int n_g = n0 + wn * (SN * 16) + sn * 16 + l16;
      float bv = 0.f;
      if (BIAS) bv = p.bias[n_g];
      bool mok = true;
      if (MASKED) mok = p.mask[(long)bz * p.N + n_g] != 0;
      #pragma unroll
      for (int r = 0; r < 8; ++r) {
        const int m_g = m0 + wm * 32 + sm * 16 + hs * 8 + r;
        float c = acc[sm][sn][r] * p.alpha + bv;
        if (RELU) c = fmaxf(c, 0.f);
        if (MASKED && !mok) c = -1.0e9f;
        const long idx = cOff + (long)m_g * p.ldc + n_g;
        if (RESIDUAL) { c += p.resid[idx]; p.resid[idx] = c; }
        if (OUT_HALF) p.outH[idx] = (half_t)c;
        else if (!RESIDUAL) p.outF[idx] = c;
      }
    }
  }
}

// ---------------------------------------------------------------------------
// Embedding gather + sinusoidal positional encoding. D=512, S=512.
// ---------------------------------------------------------------------------
__global__ void embed_kernel(const int* __restrict__ ids,
                             const float* __restrict__ emb,
                             float* __restrict__ x) {
  const int tok = blockIdx.x;          // 0..4095
  const int id  = ids[tok];
  const int s   = tok & 511;           // position within sequence (S=512)
  #pragma unroll
  for (int i = 0; i < 2; ++i) {
    const int col = threadIdx.x + i * 256;
    const int j   = (col < 256) ? col : col - 256;
    const float inv = __powf(10000.f, (2.f * (float)j) / 512.f);
    const float ang = (float)s / inv;
    const float pe  = (col < 256) ? __sinf(ang) : __cosf(ang);
    x[(long)tok * 512 + col] = emb[(long)id * 512 + col] + pe;
  }
}

// ---------------------------------------------------------------------------
// LayerNorm over D=512, one wave32 per row, output f16 for the WMMA GEMMs.
// ---------------------------------------------------------------------------
__global__ __launch_bounds__(256) void ln_kernel(const float* __restrict__ x,
                                                 const float* __restrict__ g,
                                                 const float* __restrict__ b,
                                                 half_t* __restrict__ out) {
  const long row = (long)blockIdx.x * 8 + (threadIdx.x >> 5);
  const int lane = threadIdx.x & 31;
  const float* xr = x + row * 512;
  float v[16];
  float s = 0.f;
  #pragma unroll
  for (int j = 0; j < 16; ++j) { v[j] = xr[lane + j * 32]; s += v[j]; }
  for (int off = 16; off; off >>= 1) s += __shfl_xor(s, off, 32);
  const float mu = s * (1.f / 512.f);
  float vs = 0.f;
  #pragma unroll
  for (int j = 0; j < 16; ++j) { float d = v[j] - mu; vs += d * d; }
  for (int off = 16; off; off >>= 1) vs += __shfl_xor(vs, off, 32);
  const float rstd = rsqrtf(vs * (1.f / 512.f) + 1.0e-3f);
  #pragma unroll
  for (int j = 0; j < 16; ++j) {
    const int col = lane + j * 32;
    out[row * 512 + col] = (half_t)((v[j] - mu) * rstd * g[col] + b[col]);
  }
}

// ---------------------------------------------------------------------------
// In-place softmax over rows of 512 f16 values, one wave per row (attention).
// ---------------------------------------------------------------------------
__global__ __launch_bounds__(256) void softmax512_h(half_t* __restrict__ a) {
  const long row = (long)blockIdx.x * 8 + (threadIdx.x >> 5);
  const int lane = threadIdx.x & 31;
  half_t* ar = a + row * 512;
  float v[16];
  float mx = -1.0e30f;
  #pragma unroll
  for (int j = 0; j < 16; ++j) { v[j] = (float)ar[lane + j * 32]; mx = fmaxf(mx, v[j]); }
  for (int off = 16; off; off >>= 1) mx = fmaxf(mx, __shfl_xor(mx, off, 32));
  float s = 0.f;
  #pragma unroll
  for (int j = 0; j < 16; ++j) { v[j] = __expf(v[j] - mx); s += v[j]; }
  for (int off = 16; off; off >>= 1) s += __shfl_xor(s, off, 32);
  const float inv = 1.f / s;
  #pragma unroll
  for (int j = 0; j < 16; ++j) ar[lane + j * 32] = (half_t)(v[j] * inv);
}

// ---------------------------------------------------------------------------
// In-place softmax over rows of 4096 f32 values, one block per row (logits).
// ---------------------------------------------------------------------------
__global__ __launch_bounds__(256) void softmax4096_f(float* __restrict__ a) {
  float* ar = a + (long)blockIdx.x * 4096;
  const int t = threadIdx.x;
  __shared__ float red[8];
  float v[16];
  float mx = -1.0e30f;
  #pragma unroll
  for (int j = 0; j < 16; ++j) { v[j] = ar[t + j * 256]; mx = fmaxf(mx, v[j]); }
  for (int off = 16; off; off >>= 1) mx = fmaxf(mx, __shfl_xor(mx, off, 32));
  if ((t & 31) == 0) red[t >> 5] = mx;
  __syncthreads();
  mx = -1.0e30f;
  #pragma unroll
  for (int i = 0; i < 8; ++i) mx = fmaxf(mx, red[i]);
  __syncthreads();
  float s = 0.f;
  #pragma unroll
  for (int j = 0; j < 16; ++j) { v[j] = __expf(v[j] - mx); s += v[j]; }
  for (int off = 16; off; off >>= 1) s += __shfl_xor(s, off, 32);
  if ((t & 31) == 0) red[t >> 5] = s;
  __syncthreads();
  s = 0.f;
  #pragma unroll
  for (int i = 0; i < 8; ++i) s += red[i];
  const float inv = 1.f / s;
  #pragma unroll
  for (int j = 0; j < 16; ++j) ar[t + j * 256] = v[j] * inv;
}

// f32 -> f16 conversion (element count must be a multiple of 256)
__global__ void cvt_f16(const float* __restrict__ in, half_t* __restrict__ out) {
  const long i = (long)blockIdx.x * 256 + threadIdx.x;
  out[i] = (half_t)in[i];
}

// ---------------------------------------------------------------------------
extern "C" void kernel_launch(void* const* d_in, const int* in_sizes, int n_in,
                              void* d_out, int out_size, void* d_ws, size_t ws_size,
                              hipStream_t stream) {
  (void)in_sizes; (void)n_in; (void)out_size; (void)ws_size;
  const int*           token_ids = (const int*)d_in[0];
  const unsigned char* mask      = (const unsigned char*)d_in[1];
  const float* emb   = (const float*)d_in[2];
  const float* ln1_g = (const float*)d_in[3];
  const float* ln1_b = (const float*)d_in[4];
  const float* W_Q   = (const float*)d_in[5];
  const float* W_K   = (const float*)d_in[6];
  const float* W_V   = (const float*)d_in[7];
  const float* W_O   = (const float*)d_in[8];
  const float* ln2_g = (const float*)d_in[9];
  const float* ln2_b = (const float*)d_in[10];
  const float* fW1   = (const float*)d_in[11];
  const float* fb1   = (const float*)d_in[12];
  const float* fW2   = (const float*)d_in[13];
  const float* fb2   = (const float*)d_in[14];
  const float* outW  = (const float*)d_in[15];
  const float* outB  = (const float*)d_in[16];

  constexpr int L = 6, D = 512, H = 8, F = 2048, S = 512, Bq = 8, T = 4096;
  constexpr int NT = Bq * S;      // 4096 tokens
  constexpr int dh = D / H;       // 64

  char* wp = (char*)d_ws;
  auto alloc = [&](size_t bytes) -> char* {
    char* r = wp;
    wp += (bytes + 255) & ~(size_t)255;
    return r;
  };
  float*  x      = (float*) alloc((size_t)NT * D * 4);
  half_t* h16    = (half_t*)alloc((size_t)NT * D * 2);
  half_t* q16    = (half_t*)alloc((size_t)NT * D * 2);
  half_t* k16    = (half_t*)alloc((size_t)NT * D * 2);
  half_t* v16w   = (half_t*)alloc((size_t)NT * D * 2);
  half_t* att16  = (half_t*)alloc((size_t)Bq * H * S * S * 2);
  half_t* o16    = (half_t*)alloc((size_t)NT * D * 2);
  half_t* f116   = (half_t*)alloc((size_t)NT * F * 2);
  half_t* wq16   = (half_t*)alloc((size_t)D * D * 2);
  half_t* wk16   = (half_t*)alloc((size_t)D * D * 2);
  half_t* wv16   = (half_t*)alloc((size_t)D * D * 2);
  half_t* wo16   = (half_t*)alloc((size_t)D * D * 2);
  half_t* w116   = (half_t*)alloc((size_t)D * F * 2);
  half_t* w216   = (half_t*)alloc((size_t)F * D * 2);
  half_t* wout16 = (half_t*)alloc((size_t)D * T * 2);

  embed_kernel<<<NT, 256, 0, stream>>>(token_ids, emb, x);
  cvt_f16<<<(D * T) / 256, 256, 0, stream>>>(outW, wout16);

  for (int l = 0; l < L; ++l) {
    cvt_f16<<<(D * D) / 256, 256, 0, stream>>>(W_Q + (size_t)l * D * D, wq16);
    cvt_f16<<<(D * D) / 256, 256, 0, stream>>>(W_K + (size_t)l * D * D, wk16);
    cvt_f16<<<(D * D) / 256, 256, 0, stream>>>(W_V + (size_t)l * D * D, wv16);
    cvt_f16<<<(D * D) / 256, 256, 0, stream>>>(W_O + (size_t)l * D * D, wo16);
    cvt_f16<<<(D * F) / 256, 256, 0, stream>>>(fW1 + (size_t)l * D * F, w116);
    cvt_f16<<<(F * D) / 256, 256, 0, stream>>>(fW2 + (size_t)l * F * D, w216);

    ln_kernel<<<NT / 8, 256, 0, stream>>>(x, ln1_g + l * D, ln1_b + l * D, h16);

    // ---- Q/K/V projections: (4096x512) @ (512x512) -> f16 ----------------
    GemmP p{};
    p.A = h16; p.lda = D; p.ldb = D; p.ldc = D;
    p.M = NT; p.N = D; p.K = D; p.zH = 1; p.alpha = 1.f;
    dim3 gq(D / 128, NT / 128, 1);
    p.B = wq16; p.outH = q16;
    gemm_wmma<4, false, true, false, false, false, false><<<gq, 256, 0, stream>>>(p);
    p.B = wk16; p.outH = k16;
    gemm_wmma<4, false, true, false, false, false, false><<<gq, 256, 0, stream>>>(p);
    p.B = wv16; p.outH = v16w;
    gemm_wmma<4, false, true, false, false, false, false><<<gq, 256, 0, stream>>>(p);

    // ---- scores = scale * Q @ K^T, masked, per (batch, head) -------------
    GemmP ps{};
    ps.A = q16; ps.B = k16; ps.outH = att16; ps.mask = mask;
    ps.M = S; ps.N = S; ps.K = dh;
    ps.lda = D; ps.ldb = D; ps.ldc = S;
    ps.zH = H;
    ps.aZb = (long)S * D; ps.aZh = dh;
    ps.bZb = (long)S * D; ps.bZh = dh;
    ps.cZb = (long)H * S * S; ps.cZh = (long)S * S;
    ps.alpha = 0.125f; // 1/sqrt(64)
    gemm_wmma<4, true, true, false, false, false, true>
        <<<dim3(S / 128, S / 128, Bq * H), 256, 0, stream>>>(ps);

    softmax512_h<<<(Bq * H * S) / 8, 256, 0, stream>>>(att16);

    // ---- O = attn @ V, per (batch, head), written back into (NT, D) ------
    GemmP pa{};
    pa.A = att16; pa.B = v16w; pa.outH = o16;
    pa.M = S; pa.N = dh; pa.K = S;
    pa.lda = S; pa.ldb = D; pa.ldc = D;
    pa.zH = H;
    pa.aZb = (long)H * S * S; pa.aZh = (long)S * S;
    pa.bZb = (long)S * D; pa.bZh = dh;
    pa.cZb = (long)S * D; pa.cZh = dh;
    pa.alpha = 1.f;
    gemm_wmma<2, false, true, false, false, false, false>
        <<<dim3(dh / 64, S / 128, Bq * H), 256, 0, stream>>>(pa);

    // ---- x += O @ W_O ----------------------------------------------------
    GemmP po{};
    po.A = o16; po.B = wo16; po.resid = x;
    po.M = NT; po.N = D; po.K = D;
    po.lda = D; po.ldb = D; po.ldc = D; po.zH = 1; po.alpha = 1.f;
    gemm_wmma<4, false, false, false, false, true, false>
        <<<dim3(D / 128, NT / 128, 1), 256, 0, stream>>>(po);

    ln_kernel<<<NT / 8, 256, 0, stream>>>(x, ln2_g + l * D, ln2_b + l * D, h16);

    // ---- FFN1: relu(h @ W1 + b1) -> f16 ----------------------------------
    GemmP p1{};
    p1.A = h16; p1.B = w116; p1.bias = fb1 + (size_t)l * F; p1.outH = f116;
    p1.M = NT; p1.N = F; p1.K = D;
    p1.lda = D; p1.ldb = F; p1.ldc = F; p1.zH = 1; p1.alpha = 1.f;
    gemm_wmma<4, false, true, true, true, false, false>
        <<<dim3(F / 128, NT / 128, 1), 256, 0, stream>>>(p1);

    // ---- FFN2: x += f1 @ W2 + b2 -----------------------------------------
    GemmP p2{};
    p2.A = f116; p2.B = w216; p2.bias = fb2 + (size_t)l * D; p2.resid = x;
    p2.M = NT; p2.N = D; p2.K = F;
    p2.lda = F; p2.ldb = D; p2.ldc = D; p2.zH = 1; p2.alpha = 1.f;
    gemm_wmma<4, false, false, true, false, true, false>
        <<<dim3(D / 128, NT / 128, 1), 256, 0, stream>>>(p2);
  }

  cvt_f16<<<(NT * D) / 256, 256, 0, stream>>>(x, h16);

  // ---- logits = x @ out_W + out_b (f32 into d_out), then row softmax -----
  GemmP pl{};
  pl.A = h16; pl.B = wout16; pl.bias = outB; pl.outF = (float*)d_out;
  pl.M = NT; pl.N = T; pl.K = D;
  pl.lda = D; pl.ldb = T; pl.ldc = T; pl.zH = 1; pl.alpha = 1.f;
  gemm_wmma<4, false, false, true, false, false, false>
      <<<dim3(T / 128, NT / 128, 1), 256, 0, stream>>>(pl);

  softmax4096_f<<<NT, 256, 0, stream>>>((float*)d_out);
}
